// LSTMCell_41059887349770
// MI455X (gfx1250) — matrix operand order repaired and checked by reading, not compiled, not run on hardware
//
#include <hip/hip_runtime.h>

// ---------------------------------------------------------------------------
// Fused LSTM cell for MI455X (gfx1250, wave32).
// Path A (needs 48MB workspace):
//   K1: pack x|h  -> bf16 A   [8192 x 2048]        (d_ws)
//   K2: pack Wg|Ug-> bf16 W4  [4 x 1024 x 2048]    (d_ws + 32MB)
//   K3: GEMM: bf16 tiles streamed straight to LDS via
//       global_load_async_to_lds_b128 (ASYNCcnt), double-buffered by a
//       toggled LDS byte offset (no unroll, stable accumulator registers),
//       v_wmma_f32_16x16x32_bf16 inner loop, fused sigmoid/tanh epilogue.
// Path B (fallback, no workspace): round-1 kernel (convert-in-kernel).
// ---------------------------------------------------------------------------

typedef __bf16 bf16_t;
typedef __bf16 v16bf __attribute__((ext_vector_type(16)));
typedef __bf16 v8bf  __attribute__((ext_vector_type(8)));
typedef __bf16 v4bf  __attribute__((ext_vector_type(4)));
typedef float  v8f   __attribute__((ext_vector_type(8)));
typedef float  v4f   __attribute__((ext_vector_type(4)));

#define NROWS   8192      // B
#define DDIM    1024      // D_IN == D_H
#define KDIM    2048      // packed K = D_IN + D_H
#define MT      128       // workgroup M tile
#define NT      32        // workgroup N tile (per gate)
#define KT      32        // K slice per step (one WMMA depth)
#define KSTEPS  (KDIM / KT)   // 64

// per-buffer LDS layout (bytes): [0, 8192) A tile 128x32 bf16
//                                [8192, 16384) W tiles 4 x 32x32 bf16
#define LDS_W_OFF   8192
#define LDS_BUF_STRIDE 16384

union FragAB { v16bf v; v8bf h[2]; };

__device__ __forceinline__ float fast_sigmoid(float x) {
    return 1.0f / (1.0f + __expf(-x));
}
__device__ __forceinline__ float fast_tanh(float x) {
    float t = __expf(-2.0f * fabsf(x));       // in (0,1], no overflow
    float r = (1.0f - t) / (1.0f + t);
    return copysignf(r, x);
}

// per-lane 16B async copy: global (bf16) -> LDS, tracked by ASYNCcnt
__device__ __forceinline__ void async_copy_b128(unsigned lds_addr,
                                                const bf16_t* gptr) {
    unsigned long long ga = (unsigned long long)(uintptr_t)gptr;
    asm volatile("global_load_async_to_lds_b128 %0, %1, off"
                 :: "v"(lds_addr), "v"(ga) : "memory");
}
__device__ __forceinline__ void wait_async0() {
    asm volatile("s_wait_asynccnt 0" ::: "memory");
}

// ======================= pre-pass: bf16 packing ============================

__global__ __launch_bounds__(256)
void pack_A_bf16(const float* __restrict__ x, const float* __restrict__ h,
                 bf16_t* __restrict__ A) {
    // 8 elements per thread; chunks never straddle the x|h boundary
    const size_t base = ((size_t)blockIdx.x * 256 + threadIdx.x) * 8;
    const int row = (int)(base / KDIM);
    const int k   = (int)(base % KDIM);
    const float* src = (k < DDIM) ? (x + (size_t)row * DDIM + k)
                                  : (h + (size_t)row * DDIM + (k - DDIM));
    v4f a = *(const v4f*)(src);
    v4f b = *(const v4f*)(src + 4);
    v8bf o;
    o[0]=(bf16_t)a.x; o[1]=(bf16_t)a.y; o[2]=(bf16_t)a.z; o[3]=(bf16_t)a.w;
    o[4]=(bf16_t)b.x; o[5]=(bf16_t)b.y; o[6]=(bf16_t)b.z; o[7]=(bf16_t)b.w;
    *(v8bf*)(A + base) = o;
}

__global__ __launch_bounds__(256)
void pack_W_bf16(const float* __restrict__ Wi, const float* __restrict__ Ui,
                 const float* __restrict__ Wf, const float* __restrict__ Uf,
                 const float* __restrict__ Wo, const float* __restrict__ Uo,
                 const float* __restrict__ Wc, const float* __restrict__ Uc,
                 bf16_t* __restrict__ W4) {
    const float* Wg[4] = { Wi, Wf, Wo, Wc };
    const float* Ug[4] = { Ui, Uf, Uo, Uc };
    const size_t base = ((size_t)blockIdx.x * 256 + threadIdx.x) * 8;
    const int gn = (int)(base / KDIM);      // 0..4095 : g*1024 + n
    const int g  = gn >> 10;
    const int n  = gn & 1023;
    const int k  = (int)(base % KDIM);
    const float* src = (k < DDIM) ? (Wg[g] + (size_t)n * DDIM + k)
                                  : (Ug[g] + (size_t)n * DDIM + (k - DDIM));
    v4f a = *(const v4f*)(src);
    v4f b = *(const v4f*)(src + 4);
    v8bf o;
    o[0]=(bf16_t)a.x; o[1]=(bf16_t)a.y; o[2]=(bf16_t)a.z; o[3]=(bf16_t)a.w;
    o[4]=(bf16_t)b.x; o[5]=(bf16_t)b.y; o[6]=(bf16_t)b.z; o[7]=(bf16_t)b.w;
    *(v8bf*)(W4 + base) = o;
}

// ================== main GEMM + gates (async-to-LDS path) ==================

__global__ __launch_bounds__(256, 2)
void lstm_wmma_async(const bf16_t* __restrict__ A,    // [8192][2048]
                     const bf16_t* __restrict__ W4,   // [4][1024][2048]
                     const float*  __restrict__ c,
                     const float*  __restrict__ pbi, const float* __restrict__ pbf,
                     const float*  __restrict__ pbo, const float* __restrict__ pbc,
                     float* __restrict__ out)
{
    __shared__ __align__(16) char lds[2 * LDS_BUF_STRIDE];   // 32 KB

    const int tid  = threadIdx.x;
    const int lane = tid & 31;
    const int wv   = tid >> 5;          // wave id 0..7
    const int wm   = wv >> 1;           // wave M position 0..3 (32 rows each)
    const int wn   = wv & 1;            // wave N position 0..1 (16 cols each)
    const int half = lane >> 4;         // lane group 0/1
    const int mi   = lane & 15;

    const int m0 = blockIdx.y * MT;     // row block in B
    const int n0 = blockIdx.x * NT;     // col block in H

    v8f acc[4][2];
    #pragma unroll
    for (int g = 0; g < 4; ++g)
        #pragma unroll
        for (int t = 0; t < 2; ++t)
            acc[g][t] = (v8f)(0.0f);

    // ---- async transfer indexing (16B per copy; 4 copies/thread/step) ----
    //   A tile: 512 segs: id = t*256+tid -> row = id>>2 (0..127), seg = id&3
    //   W tile: 512 segs: id = t*256+tid -> g = id>>7, r = (id>>2)&31, seg = id&3
    const int seg   = tid & 3;
    const int aRow0 = tid >> 2;              // 0..63
    const int aRow1 = 64 + (tid >> 2);       // 64..127
    const int wG0   = tid >> 7;              // 0..1
    const int wG1   = 2 + (tid >> 7);        // 2..3
    const int wR    = (tid >> 2) & 31;       // 0..31

    // LDS byte offsets (relative to current buffer base)
    const unsigned ldsA0 = (unsigned)((aRow0 * KT + seg * 8) * 2);
    const unsigned ldsA1 = (unsigned)((aRow1 * KT + seg * 8) * 2);
    const unsigned ldsW0 = (unsigned)(LDS_W_OFF + ((wG0 * NT + wR) * KT + seg * 8) * 2);
    const unsigned ldsW1 = (unsigned)(LDS_W_OFF + ((wG1 * NT + wR) * KT + seg * 8) * 2);
    const unsigned ldsBase = (unsigned)(uintptr_t)lds;

    // running global source pointers (advance by KT per step)
    const bf16_t* pA0 = A  + (size_t)(m0 + aRow0) * KDIM + seg * 8;
    const bf16_t* pA1 = A  + (size_t)(m0 + aRow1) * KDIM + seg * 8;
    const bf16_t* pW0 = W4 + ((size_t)wG0 * DDIM + n0 + wR) * KDIM + seg * 8;
    const bf16_t* pW1 = W4 + ((size_t)wG1 * DDIM + n0 + wR) * KDIM + seg * 8;

    // fragment LDS byte offsets (relative to current buffer base)
    const unsigned fragA0 = (unsigned)(((wm * 32 + mi) * KT) * 2);          // + t*16*KT*2
    const unsigned fragW0 = (unsigned)(LDS_W_OFF + ((wn * 16 + mi) * KT + half * 8) * 2);

    // ---- preload step 0 into buffer 0 ----
    async_copy_b128(ldsBase + ldsA0, pA0);  pA0 += KT;
    async_copy_b128(ldsBase + ldsA1, pA1);  pA1 += KT;
    async_copy_b128(ldsBase + ldsW0, pW0);  pW0 += KT;
    async_copy_b128(ldsBase + ldsW1, pW1);  pW1 += KT;

    unsigned curOff = 0;

    #pragma unroll 1
    for (int ks = 0; ks < KSTEPS; ++ks) {
        wait_async0();        // this wave's 4 copies for step ks are in LDS
        __syncthreads();      // everyone's copies are in LDS; prior reads done

        const unsigned nxtOff = curOff ^ LDS_BUF_STRIDE;

        // ---- issue async copies for step ks+1 into the other buffer ----
        if (ks + 1 < KSTEPS) {
            async_copy_b128(ldsBase + nxtOff + ldsA0, pA0);  pA0 += KT;
            async_copy_b128(ldsBase + nxtOff + ldsA1, pA1);  pA1 += KT;
            async_copy_b128(ldsBase + nxtOff + ldsW0, pW0);  pW0 += KT;
            async_copy_b128(ldsBase + nxtOff + ldsW1, pW1);  pW1 += KT;
        }

        // ---- load fragments from LDS (ISA 16-bit A/B layouts) ----
        const char* bufp = lds + curOff;
        FragAB a[2];
        #pragma unroll
        for (int t = 0; t < 2; ++t) {
            const bf16_t* base = (const bf16_t*)(bufp + fragA0 + t * (16 * KT * 2));
            a[t].h[0] = *(const v8bf*)(base + half * 8);        // K 0..7  / 8..15
            a[t].h[1] = *(const v8bf*)(base + 16 + half * 8);   // K 16..23/ 24..31
        }
        FragAB w[4];
        #pragma unroll
        for (int g = 0; g < 4; ++g) {
            const bf16_t* base = (const bf16_t*)(bufp + fragW0 + g * (NT * KT * 2));
            w[g].h[0] = *(const v8bf*)(base);                   // K 0..7  / 8..15
            w[g].h[1] = *(const v8bf*)(base + 16);              // K 16..23/ 24..31
        }

        // ---- 8 WMMAs per wave per K-step ----
        #pragma unroll
        for (int g = 0; g < 4; ++g)
            #pragma unroll
            for (int t = 0; t < 2; ++t)
                acc[g][t] = __builtin_amdgcn_wmma_f32_16x16x32_bf16(
                    false, a[t].v, false, w[g].v, (short)0, acc[g][t], false, false);

        curOff = nxtOff;
    }

    // ---------------- fused gate epilogue ----------------
    const int colH = n0 + wn * 16 + mi;
    const float bI = pbi[colH];
    const float bF = pbf[colH];
    const float bO = pbo[colH];
    const float bC = pbc[colH];

    float* outH = out;
    float* outC = out + (size_t)NROWS * DDIM;

    #pragma unroll
    for (int t = 0; t < 2; ++t) {
        #pragma unroll
        for (int v = 0; v < 8; ++v) {
            const int row = m0 + wm * 32 + t * 16 + v + 8 * half;  // C layout
            const size_t off = (size_t)row * DDIM + colH;
            const float gi = acc[0][t][v] + bI;
            const float gf = acc[1][t][v] + bF;
            const float go = acc[2][t][v] + bO;
            const float gc = acc[3][t][v] + bC;
            const float i_ = fast_sigmoid(gi);
            const float f_ = fast_sigmoid(gf);
            const float o_ = fast_sigmoid(go);
            const float cb = fast_tanh(gc);
            const float nc = f_ * c[off] + i_ * cb;
            const float nh = o_ * fast_tanh(nc);
            outH[off] = nh;
            outC[off] = nc;
        }
    }
}

// =============== fallback: round-1 convert-in-kernel version ===============

__global__ __launch_bounds__(256, 2)
void lstm_wmma_fb(const float* __restrict__ x,
                  const float* __restrict__ h,
                  const float* __restrict__ c,
                  const float* __restrict__ Wi, const float* __restrict__ Ui,
                  const float* __restrict__ Wf, const float* __restrict__ Uf,
                  const float* __restrict__ Wo, const float* __restrict__ Uo,
                  const float* __restrict__ Wc, const float* __restrict__ Uc,
                  const float* __restrict__ pbi, const float* __restrict__ pbf,
                  const float* __restrict__ pbo, const float* __restrict__ pbc,
                  float* __restrict__ out)
{
    __shared__ bf16_t sA[2][MT * KT];
    __shared__ bf16_t sW[2][4][NT * KT];

    const float* Wg[4] = { Wi, Wf, Wo, Wc };
    const float* Ug[4] = { Ui, Uf, Uo, Uc };

    const int tid  = threadIdx.x;
    const int lane = tid & 31;
    const int wv   = tid >> 5;
    const int wm   = wv >> 1;
    const int wn   = wv & 1;
    const int half = lane >> 4;
    const int mi   = lane & 15;

    const int m0 = blockIdx.y * MT;
    const int n0 = blockIdx.x * NT;

    v8f acc[4][2];
    #pragma unroll
    for (int g = 0; g < 4; ++g)
        #pragma unroll
        for (int t = 0; t < 2; ++t)
            acc[g][t] = (v8f)(0.0f);

    v4f ga[4];
    v4f gw[4];

    {
        #pragma unroll
        for (int it = 0; it < 4; ++it) {
            int idx = it * 256 + tid;
            int r = idx >> 3, cv = idx & 7;
            ga[it] = *(const v4f*)(x + (size_t)(m0 + r) * DDIM + cv * 4);
        }
        int r = tid >> 3, cv = tid & 7;
        #pragma unroll
        for (int g = 0; g < 4; ++g)
            gw[g] = *(const v4f*)(Wg[g] + (size_t)(n0 + r) * DDIM + cv * 4);
    }

    for (int ks = 0; ks < KSTEPS; ++ks) {
        const int buf = ks & 1;

        #pragma unroll
        for (int it = 0; it < 4; ++it) {
            int idx = it * 256 + tid;
            int r = idx >> 3, cv = idx & 7;
            v4bf pk;
            pk.x = (bf16_t)ga[it].x; pk.y = (bf16_t)ga[it].y;
            pk.z = (bf16_t)ga[it].z; pk.w = (bf16_t)ga[it].w;
            *(v4bf*)&sA[buf][r * KT + cv * 4] = pk;
        }
        {
            int r = tid >> 3, cv = tid & 7;
            #pragma unroll
            for (int g = 0; g < 4; ++g) {
                v4bf pk;
                pk.x = (bf16_t)gw[g].x; pk.y = (bf16_t)gw[g].y;
                pk.z = (bf16_t)gw[g].z; pk.w = (bf16_t)gw[g].w;
                *(v4bf*)&sW[buf][g][r * KT + cv * 4] = pk;
            }
        }

        if (ks + 1 < KSTEPS) {
            const int nk = ks + 1;
            const float* Asrc = (nk < KSTEPS / 2) ? x : h;
            const int k0 = (nk & 31) * KT;
            #pragma unroll
            for (int it = 0; it < 4; ++it) {
                int idx = it * 256 + tid;
                int r = idx >> 3, cv = idx & 7;
                ga[it] = *(const v4f*)(Asrc + (size_t)(m0 + r) * DDIM + k0 + cv * 4);
            }
            int r = tid >> 3, cv = tid & 7;
            #pragma unroll
            for (int g = 0; g < 4; ++g) {
                const float* Ws = (nk < KSTEPS / 2) ? Wg[g] : Ug[g];
                gw[g] = *(const v4f*)(Ws + (size_t)(n0 + r) * DDIM + k0 + cv * 4);
            }
        }

        __syncthreads();

        FragAB a[2];
        #pragma unroll
        for (int t = 0; t < 2; ++t) {
            const bf16_t* base = &sA[buf][(wm * 32 + t * 16 + mi) * KT];
            a[t].h[0] = *(const v8bf*)(base + half * 8);
            a[t].h[1] = *(const v8bf*)(base + 16 + half * 8);
        }
        FragAB w[4];
        #pragma unroll
        for (int g = 0; g < 4; ++g) {
            const bf16_t* base = &sW[buf][g][(wn * 16 + mi) * KT + half * 8];
            w[g].h[0] = *(const v8bf*)(base);
            w[g].h[1] = *(const v8bf*)(base + 16);
        }

        #pragma unroll
        for (int g = 0; g < 4; ++g)
            #pragma unroll
            for (int t = 0; t < 2; ++t)
                acc[g][t] = __builtin_amdgcn_wmma_f32_16x16x32_bf16(
                    false, a[t].v, false, w[g].v, (short)0, acc[g][t], false, false);
    }

    const int colH = n0 + wn * 16 + mi;
    const float bI = pbi[colH];
    const float bF = pbf[colH];
    const float bO = pbo[colH];
    const float bC = pbc[colH];

    float* outH = out;
    float* outC = out + (size_t)NROWS * DDIM;

    #pragma unroll
    for (int t = 0; t < 2; ++t) {
        #pragma unroll
        for (int v = 0; v < 8; ++v) {
            const int row = m0 + wm * 32 + t * 16 + v + 8 * half;
            const size_t off = (size_t)row * DDIM + colH;
            const float gi = acc[0][t][v] + bI;
            const float gf = acc[1][t][v] + bF;
            const float go = acc[2][t][v] + bO;
            const float gc = acc[3][t][v] + bC;
            const float i_ = fast_sigmoid(gi);
            const float f_ = fast_sigmoid(gf);
            const float o_ = fast_sigmoid(go);
            const float cb = fast_tanh(gc);
            const float nc = f_ * c[off] + i_ * cb;
            const float nh = o_ * fast_tanh(nc);
            outH[off] = nh;
            outC[off] = nc;
        }
    }
}

// ============================== launcher ===================================

extern "C" void kernel_launch(void* const* d_in, const int* in_sizes, int n_in,
                              void* d_out, int out_size, void* d_ws, size_t ws_size,
                              hipStream_t stream) {
    (void)in_sizes; (void)n_in; (void)out_size;
    const float* x  = (const float*)d_in[0];
    const float* h  = (const float*)d_in[1];
    const float* c  = (const float*)d_in[2];
    const float* Wi = (const float*)d_in[3];
    const float* bi = (const float*)d_in[4];
    const float* Ui = (const float*)d_in[5];
    const float* Wf = (const float*)d_in[6];
    const float* bf = (const float*)d_in[7];
    const float* Uf = (const float*)d_in[8];
    const float* Wo = (const float*)d_in[9];
    const float* bo = (const float*)d_in[10];
    const float* Uo = (const float*)d_in[11];
    const float* Wc = (const float*)d_in[12];
    const float* bc = (const float*)d_in[13];
    const float* Uc = (const float*)d_in[14];
    float* out = (float*)d_out;

    const size_t A_BYTES = (size_t)NROWS * KDIM * sizeof(bf16_t);      // 32 MB
    const size_t W_BYTES = (size_t)4 * DDIM * KDIM * sizeof(bf16_t);   // 16 MB

    dim3 grid(DDIM / NT, NROWS / MT);   // 32 x 64
    dim3 block(256);

    if (d_ws && ws_size >= A_BYTES + W_BYTES) {
        bf16_t* Abf = (bf16_t*)d_ws;
        bf16_t* W4  = (bf16_t*)((char*)d_ws + A_BYTES);
        hipLaunchKernelGGL(pack_A_bf16,
                           dim3((unsigned)((size_t)NROWS * KDIM / (8 * 256))), dim3(256),
                           0, stream, x, h, Abf);
        hipLaunchKernelGGL(pack_W_bf16,
                           dim3((unsigned)((size_t)4 * DDIM * KDIM / (8 * 256))), dim3(256),
                           0, stream, Wi, Ui, Wf, Uf, Wo, Uo, Wc, Uc, W4);
        hipLaunchKernelGGL(lstm_wmma_async, grid, block, 0, stream,
                           Abf, W4, c, bi, bf, bo, bc, out);
    } else {
        hipLaunchKernelGGL(lstm_wmma_fb, grid, block, 0, stream,
                           x, h, c, Wi, Ui, Wf, Uf, Wo, Uo, Wc, Uc,
                           bi, bf, bo, bc, out);
    }
}
